// myNN_9818295238997
// MI455X (gfx1250) — matrix operand order repaired and checked by reading, not compiled, not run on hardware
//
#include <hip/hip_runtime.h>
#include <hip/hip_bf16.h>

#define EMBED 64
#define TSEQ  256
#define BATCH 32
#define MROWS (BATCH*TSEQ)      // 8192
#define VOC   50257
#define NT    3142              // ceil(50257/16) n-tiles

typedef __attribute__((ext_vector_type(16))) _Float16 v16h;
typedef __attribute__((ext_vector_type(8)))  _Float16 v8h;
typedef __attribute__((ext_vector_type(8)))  float    v8f;

// ---------------------------------------------------------------------------
// K1: repack fc_w (f32 [64, VOC]) into f16 WMMA-B layout.
// bp[(((t*2)+kk)*32 + lane)*16 + h] = fc_w[32*kk + 8*(lane>>4) + (h&7) + 16*(h>>3)][16*t + (lane&15)]
// so each lane's 16 halves for one WMMA are one contiguous 32B chunk.
// ---------------------------------------------------------------------------
__global__ void k_pack_b(const float* __restrict__ fc_w, _Float16* __restrict__ bp) {
    const int TOTD = NT * 2 * 32 * 8;              // dwords (pairs of halves)
    int idx = blockIdx.x * 256 + threadIdx.x;
    if (idx >= TOTD) return;
    int v    = idx & 7;
    int lane = (idx >> 3) & 31;
    int kk   = (idx >> 8) & 1;
    int t    = idx >> 9;
    int k0 = 32 * kk + 8 * (lane >> 4) + 2 * (v & 3) + 16 * (v >> 2);
    int n  = 16 * t + (lane & 15);
    float f0 = 0.f, f1 = 0.f;
    if (n < VOC) { f0 = fc_w[k0 * VOC + n]; f1 = fc_w[(k0 + 1) * VOC + n]; }
    bp[idx * 2]     = (_Float16)f0;
    bp[idx * 2 + 1] = (_Float16)f1;
}

// ---------------------------------------------------------------------------
// K2: h = embed_w[x] + pos_w   (elementwise, [8192,64])
// ---------------------------------------------------------------------------
__global__ void k_embed(const int* __restrict__ x, const float* __restrict__ ew,
                        const float* __restrict__ pw, float* __restrict__ h) {
    int gid = blockIdx.x * 256 + threadIdx.x;      // < MROWS*64
    int m = gid >> 6, e = gid & 63;
    int tok = x[m];
    h[gid] = ew[tok * 64 + e] + pw[(m & 255) * 64 + e];
}

// ---------------------------------------------------------------------------
// K3: row LayerNorm (thread per row)
// ---------------------------------------------------------------------------
__global__ void k_ln(const float* __restrict__ h, const float* __restrict__ g,
                     const float* __restrict__ b, float* __restrict__ hn) {
    int row = blockIdx.x * 256 + threadIdx.x;
    const float* hr = h + row * 64;
    float mu = 0.f;
    for (int e = 0; e < 64; e++) mu += hr[e];
    mu *= (1.f / 64.f);
    float var = 0.f;
    for (int e = 0; e < 64; e++) { float d = hr[e] - mu; var += d * d; }
    var *= (1.f / 64.f);
    float rs = rsqrtf(var + 1e-5f);
    for (int e = 0; e < 64; e++) hn[row * 64 + e] = (hr[e] - mu) * rs * g[e] + b[e];
}

// ---------------------------------------------------------------------------
// K4: Q,K,V = hn @ Wq/Wk/Wv  (64x64), weights staged in LDS
// ---------------------------------------------------------------------------
__global__ void k_qkv(const float* __restrict__ hn, const float* __restrict__ Wq,
                      const float* __restrict__ Wk, const float* __restrict__ Wv,
                      float* __restrict__ Q, float* __restrict__ K, float* __restrict__ V) {
    __shared__ float sW[3][64 * 64];
    for (int i = threadIdx.x; i < 64 * 64; i += 256) {
        sW[0][i] = Wq[i]; sW[1][i] = Wk[i]; sW[2][i] = Wv[i];
    }
    __syncthreads();
    int gid = blockIdx.x * 256 + threadIdx.x;
    int m = gid >> 6, c = gid & 63;
    float aq = 0.f, ak = 0.f, av = 0.f;
    for (int e = 0; e < 64; e++) {
        float hv = hn[m * 64 + e];
        aq += hv * sW[0][e * 64 + c];
        ak += hv * sW[1][e * 64 + c];
        av += hv * sW[2][e * 64 + c];
    }
    Q[gid] = aq; K[gid] = ak; V[gid] = av;
}

// ---------------------------------------------------------------------------
// K5: causal attention (online softmax per query row) + residual + final LN
//     -> emits f16 A-matrix for the logits WMMA GEMM. One block per batch.
// ---------------------------------------------------------------------------
__global__ void k_attn(const float* __restrict__ Q, const float* __restrict__ K,
                       const float* __restrict__ V, const float* __restrict__ h,
                       const float* __restrict__ lg, const float* __restrict__ lb,
                       _Float16* __restrict__ aout) {
    __shared__ float sK[TSEQ * 64];
    __shared__ float sV[TSEQ * 64];
    int b = blockIdx.x;
    int i = threadIdx.x;                           // query row within batch
    int base = b * TSEQ * 64;
    for (int idx = threadIdx.x; idx < TSEQ * 64; idx += 256) {
        sK[idx] = K[base + idx];
        sV[idx] = V[base + idx];
    }
    __syncthreads();

    float q[64];
    #pragma unroll
    for (int e = 0; e < 64; e++) q[e] = Q[base + i * 64 + e];

    float acc[64];
    #pragma unroll
    for (int e = 0; e < 64; e++) acc[e] = 0.f;

    float mrun = -3.4e38f, l = 0.f;
    const float scale = 0.125f;                    // 1/sqrt(64)
    for (int j = 0; j <= i; j++) {                 // causal; sK[j]/sV[j] broadcast reads
        float s = 0.f;
        #pragma unroll
        for (int e = 0; e < 64; e++) s += q[e] * sK[j * 64 + e];
        s *= scale;
        float mnew = fmaxf(mrun, s);
        float corr = __expf(mrun - mnew);
        float w    = __expf(s - mnew);
        l = l * corr + w;
        #pragma unroll
        for (int e = 0; e < 64; e++) acc[e] = acc[e] * corr + w * sV[j * 64 + e];
        mrun = mnew;
    }
    float inv = 1.f / l;

    // residual + final layernorm, reuse acc[] as the fused row
    float mu = 0.f;
    #pragma unroll
    for (int e = 0; e < 64; e++) { acc[e] = h[base + i * 64 + e] + acc[e] * inv; mu += acc[e]; }
    mu *= (1.f / 64.f);
    float var = 0.f;
    #pragma unroll
    for (int e = 0; e < 64; e++) { float d = acc[e] - mu; var += d * d; }
    var *= (1.f / 64.f);
    float rs = rsqrtf(var + 1e-5f);
    #pragma unroll
    for (int e = 0; e < 64; e++)
        aout[base + i * 64 + e] = (_Float16)((acc[e] - mu) * rs * lg[e] + lb[e]);
}

// ---------------------------------------------------------------------------
// K6: logits = A[8192,64](f16) @ fc_w + fc_b via v_wmma_f32_16x16x32_f16.
// 8 waves/block; each wave owns a 32-row M-slab (two 16x16 M-tiles sharing B)
// and 4 N-tiles. A layout per ISA 7.12.2 (16-bit A 16x32): lane half selects
// K {0-7,16-23} vs {8-15,24-31} -> two b128 loads per K-32 chunk. B is
// pre-packed so each WMMA B operand is one contiguous 32B chunk per lane.
// C/D layout: lane L, vgpr r -> (M = r + 8*(L>>4), N = L&15).
// Output stores are non-temporal: 1.65 GB write-once stream must not rinse
// the L2-resident A (1 MB) and B (6.4 MB) operands.
// ---------------------------------------------------------------------------
__global__ void __launch_bounds__(256) k_logits(const _Float16* __restrict__ A,
                                                const _Float16* __restrict__ Bp,
                                                const float* __restrict__ bias,
                                                float* __restrict__ out) {
    int lane = threadIdx.x & 31;
    int wave = threadIdx.x >> 5;                   // 0..7
    int m0 = blockIdx.x * 32;                      // 32-row slab per block
    int tbase = blockIdx.y * 32 + wave * 4;
    int half = lane >> 4;
    int mrow = m0 + (lane & 15);

    // A for M-tile 0 (rows m0..m0+15) and M-tile 1 (rows m0+16..m0+31)
    const _Float16* ap0 = A + mrow * 64 + half * 8;
    const _Float16* ap1 = ap0 + 16 * 64;
    v8h a0lo = *(const v8h*)(ap0);                 // K  0..7   /  8..15
    v8h a0hi = *(const v8h*)(ap0 + 16);            // K 16..23  / 24..31
    v8h a1lo = *(const v8h*)(ap0 + 32);            // K 32..39  / 40..47
    v8h a1hi = *(const v8h*)(ap0 + 48);            // K 48..55  / 56..63
    v8h b0lo = *(const v8h*)(ap1);
    v8h b0hi = *(const v8h*)(ap1 + 16);
    v8h b1lo = *(const v8h*)(ap1 + 32);
    v8h b1hi = *(const v8h*)(ap1 + 48);
    v16h a0, a1, a2, a3;
    #pragma unroll
    for (int i = 0; i < 8; i++) {
        a0[i] = a0lo[i]; a0[8 + i] = a0hi[i];
        a1[i] = a1lo[i]; a1[8 + i] = a1hi[i];
        a2[i] = b0lo[i]; a2[8 + i] = b0hi[i];
        a3[i] = b1lo[i]; a3[8 + i] = b1hi[i];
    }

    for (int qi = 0; qi < 4; qi++) {
        int t = tbase + qi;
        if (t >= NT) break;                        // wave-uniform
        const v16h* bpt = (const v16h*)(Bp) + (size_t)(t * 2) * 32 + lane;
        v16h bk0 = bpt[0];                         // K 0..31  tile
        v16h bk1 = bpt[32];                        // K 32..63 tile
        v8f c0 = {}, c1 = {};
        c0 = __builtin_amdgcn_wmma_f32_16x16x32_f16(false, a0, false, bk0, (short)0, c0, false, false);
        c0 = __builtin_amdgcn_wmma_f32_16x16x32_f16(false, a1, false, bk1, (short)0, c0, false, false);
        c1 = __builtin_amdgcn_wmma_f32_16x16x32_f16(false, a2, false, bk0, (short)0, c1, false, false);
        c1 = __builtin_amdgcn_wmma_f32_16x16x32_f16(false, a3, false, bk1, (short)0, c1, false, false);
        int n = t * 16 + (lane & 15);
        if (n < VOC) {
            float bs = bias[n];
            float* o0 = out + (size_t)(m0 + 8 * half) * VOC + n;
            #pragma unroll
            for (int r = 0; r < 8; r++)
                __builtin_nontemporal_store(c0[r] + bs, o0 + (size_t)r * VOC);
            float* o1 = o0 + (size_t)16 * VOC;
            #pragma unroll
            for (int r = 0; r < 8; r++)
                __builtin_nontemporal_store(c1[r] + bs, o1 + (size_t)r * VOC);
        }
    }
}

// ---------------------------------------------------------------------------
extern "C" void kernel_launch(void* const* d_in, const int* in_sizes, int n_in,
                              void* d_out, int out_size, void* d_ws, size_t ws_size,
                              hipStream_t stream) {
    const int*   x      = (const int*)  d_in[0];
    const float* ew     = (const float*)d_in[1];
    const float* pw     = (const float*)d_in[2];
    const float* ln1_g  = (const float*)d_in[3];
    const float* ln1_b  = (const float*)d_in[4];
    const float* Wq     = (const float*)d_in[5];
    const float* Wk     = (const float*)d_in[6];
    const float* Wv     = (const float*)d_in[7];
    const float* lnf_g  = (const float*)d_in[8];
    const float* lnf_b  = (const float*)d_in[9];
    const float* fc_w   = (const float*)d_in[10];
    const float* fc_b   = (const float*)d_in[11];
    float* out = (float*)d_out;

    // workspace carve-up (all chunk sizes 256B-aligned)
    char* w = (char*)d_ws;
    _Float16* bp = (_Float16*)w;            w += (size_t)NT * 2 * 32 * 16 * 2;  // 6.43 MB
    float* h  = (float*)w;                  w += (size_t)MROWS * 64 * 4;        // 2 MB
    float* hn = (float*)w;                  w += (size_t)MROWS * 64 * 4;        // 2 MB
    float* Qb = (float*)w;                  w += (size_t)MROWS * 64 * 4;        // 2 MB
    float* Kb = (float*)w;                  w += (size_t)MROWS * 64 * 4;        // 2 MB
    float* Vb = (float*)w;                  w += (size_t)MROWS * 64 * 4;        // 2 MB
    _Float16* af = (_Float16*)w;            w += (size_t)MROWS * 64 * 2;        // 1 MB
    (void)ws_size; (void)in_sizes; (void)n_in; (void)out_size;

    const int TOTD = NT * 2 * 32 * 8;
    k_pack_b<<<(TOTD + 255) / 256, 256, 0, stream>>>(fc_w, bp);
    k_embed <<<(MROWS * 64) / 256, 256, 0, stream>>>(x, ew, pw, h);
    k_ln    <<<MROWS / 256, 256, 0, stream>>>(h, ln1_g, ln1_b, hn);
    k_qkv   <<<(MROWS * 64) / 256, 256, 0, stream>>>(hn, Wq, Wk, Wv, Qb, Kb, Vb);
    k_attn  <<<BATCH, 256, 0, stream>>>(Qb, Kb, Vb, h, lnf_g, lnf_b, af);

    dim3 grid((MROWS / 32), (NT + 31) / 32);       // 256 x 99
    k_logits<<<grid, 256, 0, stream>>>(af, bp, fc_b, out);
}
